// TreePool_65841848648314
// MI455X (gfx1250) — matrix operand order repaired and checked by reading, not compiled, not run on hardware
//
#include <hip/hip_runtime.h>
#include <hip/hip_bf16.h>

typedef __attribute__((ext_vector_type(16))) __bf16       bf16x16;
typedef __attribute__((ext_vector_type(8)))  float        f32x8;
typedef __attribute__((ext_vector_type(4)))  float        f32x4;
typedef __attribute__((ext_vector_type(4)))  unsigned int u32x4;
typedef __attribute__((ext_vector_type(8)))  unsigned int u32x8;

#define N_NODES 65535
#define DIN 288
#define G4  1024
#define HIDN 256
#define BIGF 1e30f

__device__ __forceinline__ float sigmoidf_(float x) { return 1.f / (1.f + __expf(-x)); }

// Load a 32-byte WMMA operand fragment as two 16B vector loads (both slices are
// 16B-aligned by construction), bit-cast to v16bf with no half-register repacking.
__device__ __forceinline__ bf16x16 ld_frag(const __bf16* p0, const __bf16* p1) {
  const u32x4 lo = *(const u32x4*)p0;
  const u32x4 hi = *(const u32x4*)p1;
  return __builtin_bit_cast(bf16x16, __builtin_shufflevector(lo, hi, 0, 1, 2, 3, 4, 5, 6, 7));
}

// 16 f32 -> v16bf (compiler emits v_cvt_pk_bf16_f32 pairs)
__device__ __forceinline__ bf16x16 cvt_frag16(const f32x4 a, const f32x4 b,
                                              const f32x4 c, const f32x4 d) {
  bf16x16 v;
#pragma unroll
  for (int i = 0; i < 4; ++i) {
    v[i]      = (__bf16)a[i];
    v[4 + i]  = (__bf16)b[i];
    v[8 + i]  = (__bf16)c[i];
    v[12 + i] = (__bf16)d[i];
  }
  return v;
}

// ---------------------------------------------------------------------------
// One 16x16 tile of  out += A(rowBase.., :K) @ W^T(colBase.., :K)  via
// v_wmma_f32_16x16x32_bf16.  Fragment layouts per CDNA5 ISA 7.12.2 (wave32):
//   A: lane(0..15)=row M, holds K {8h..8h+7} U {16+8h..16+8h+7}, h=lane>>4
//   B: lane(0..15)=col N, holds K in the same order as A
//   D: lane=col, VGPR v holds row M = v + 8h
// Requires: Ktot % 32 == 0, A/W rows 16B-aligned.
// ---------------------------------------------------------------------------
__device__ __forceinline__ f32x8 gemm_tile_bf16A(
    const __bf16* __restrict__ A, long lda, long rowBase, long rowCount,
    const __bf16* __restrict__ W, long ldw, long colBase,
    int Ktot, f32x8 acc)
{
  const int lane = threadIdx.x & 31;
  const int half = lane >> 4;
  const int r15  = lane & 15;
  long ar = rowBase + r15; if (ar >= rowCount) ar = rowCount - 1;   // clamp (masked on store)
  const __bf16* Arow = A + ar * lda;
  const __bf16* Wrow = W + (colBase + r15) * ldw;
#pragma unroll 4
  for (int k0 = 0; k0 < Ktot; k0 += 32) {
    const bf16x16 av = ld_frag(Arow + k0 + 8 * half, Arow + k0 + 16 + 8 * half);
    const bf16x16 bv = ld_frag(Wrow + k0 + 8 * half, Wrow + k0 + 16 + 8 * half);
    acc = __builtin_amdgcn_wmma_f32_16x16x32_bf16(false, av, false, bv, (short)0, acc, false, false);
  }
  return acc;
}

// A is f32 in memory (converted on load); Ktot may have a non-multiple-of-32
// tail (zero-padded, guarded only in the remainder tile).
__device__ __forceinline__ f32x8 gemm_tile_f32A(
    const float* __restrict__ A, long lda, long rowBase, long rowCount,
    const __bf16* __restrict__ W, long ldw, long colBase,
    int Ktot, f32x8 acc)
{
  const int lane = threadIdx.x & 31;
  const int half = lane >> 4;
  const int r15  = lane & 15;
  long ar = rowBase + r15; if (ar >= rowCount) ar = rowCount - 1;
  const float*  Arow = A + ar * lda;
  const __bf16* Wrow = W + (colBase + r15) * ldw;
  const int Kfull = Ktot & ~31;
#pragma unroll 2
  for (int k0 = 0; k0 < Kfull; k0 += 32) {
    const f32x4 a = *(const f32x4*)(Arow + k0 + 8 * half);
    const f32x4 b = *(const f32x4*)(Arow + k0 + 8 * half + 4);
    const f32x4 c = *(const f32x4*)(Arow + k0 + 16 + 8 * half);
    const f32x4 d = *(const f32x4*)(Arow + k0 + 16 + 8 * half + 4);
    const bf16x16 av = cvt_frag16(a, b, c, d);
    const bf16x16 bv = ld_frag(Wrow + k0 + 8 * half, Wrow + k0 + 16 + 8 * half);
    acc = __builtin_amdgcn_wmma_f32_16x16x32_bf16(false, av, false, bv, (short)0, acc, false, false);
  }
  if (Kfull < Ktot) {                       // guarded zero-padded tail tile
    const int k0 = Kfull;
    bf16x16 av, bv;
#pragma unroll
    for (int j = 0; j < 8; ++j) {
      const int ka = k0 + ((j < 4) ? (8 * half + 2 * j) : (16 + 8 * half + 2 * (j - 4)));
      const int kb = ka;
      av[2 * j]     = (ka     < Ktot) ? (__bf16)Arow[ka]     : (__bf16)0.f;
      av[2 * j + 1] = (ka + 1 < Ktot) ? (__bf16)Arow[ka + 1] : (__bf16)0.f;
      bv[2 * j]     = (kb     < Ktot) ? Wrow[kb]             : (__bf16)0.f;
      bv[2 * j + 1] = (kb + 1 < Ktot) ? Wrow[kb + 1]         : (__bf16)0.f;
    }
    acc = __builtin_amdgcn_wmma_f32_16x16x32_bf16(false, av, false, bv, (short)0, acc, false, false);
  }
  return acc;
}

// ---------------------------------------------------------------------------
// Kernel 0: one-shot f32 -> bf16 weight conversion into workspace
// ---------------------------------------------------------------------------
__global__ void cvt_weights_k(const float* wp, const float* wb, const float* wi, const float* wh,
                              __bf16* wpb, __bf16* wbb, __bf16* wib, __bf16* whb) {
  const int i = blockIdx.x * blockDim.x + threadIdx.x;
  if (i < 64 * 64)     wpb[i] = (__bf16)wp[i];
  if (i < 128 * 1000)  wbb[i] = (__bf16)wb[i];
  if (i < 1024 * 288)  wib[i] = (__bf16)wi[i];
  if (i < 1024 * 256)  whb[i] = (__bf16)wh[i];
}

// ---------------------------------------------------------------------------
// Kernel 1: x[:, 0:32] = op_vec @ W_op.T + b_op   (tiny 32x32, plain VALU)
// ---------------------------------------------------------------------------
__global__ void op_feat_k(const float* __restrict__ op_vec, const float* __restrict__ W_op,
                          const float* __restrict__ b_op, __bf16* __restrict__ xbf) {
  const long i = (long)blockIdx.x * blockDim.x + threadIdx.x;
  if (i >= (long)N_NODES * 32) return;
  const long node = i >> 5;
  const int  j    = (int)(i & 31);
  const float* r = op_vec + node * 32;
  const float* w = W_op + j * 32;
  float acc = b_op[j];
#pragma unroll
  for (int k = 0; k < 32; ++k) acc += r[k] * w[k];
  xbf[node * DIN + j] = (__bf16)acc;
}

// ---------------------------------------------------------------------------
// Kernel 2: masked min/max pooling of  preds @ W_pred.T + b_pred  over K=8.
// One wave = one 16-row strip = 2 nodes.  Mask fetched once as a u64 (8 bytes),
// pooling is branchless v_cndmask selects.
// ---------------------------------------------------------------------------
__global__ void pred_pool_k(const float* __restrict__ preds,
                            const unsigned char* __restrict__ mask,
                            const unsigned char* __restrict__ is_and,
                            const __bf16* __restrict__ Wpred,
                            const float* __restrict__ bpred,
                            __bf16* __restrict__ xbf, int condOfs) {
  const int wave = threadIdx.x >> 5;
  const int lane = threadIdx.x & 31;
  const int half = lane >> 4, r15 = lane & 15;
  const long strip   = (long)blockIdx.x * 8 + wave;
  const long nRows   = (long)N_NODES * 8;
  const long rowBase = strip * 16;
  const int  node    = (int)(strip * 2 + half);
  const bool nodeOk  = node < N_NODES;
  const int  nodeC   = nodeOk ? node : (N_NODES - 1);   // clamp: loads unconditional
  const unsigned long long mb8 = *(const unsigned long long*)(mask + (long)nodeC * 8);
  const int andv = is_and[nodeC] != 0;
#pragma unroll
  for (int t = 0; t < 4; ++t) {
    f32x8 acc = {0.f, 0.f, 0.f, 0.f, 0.f, 0.f, 0.f, 0.f};
    acc = gemm_tile_f32A(preds, 64, rowBase, nRows, Wpred, 64, t * 16, 64, acc);
    const float bias = bpred[t * 16 + r15];
    float mn = BIGF, mx = -BIGF;
#pragma unroll
    for (int v = 0; v < 8; ++v) {
      const float e = acc[v] + bias;
      const bool mbit = ((mb8 >> (8 * v)) & 0xffull) != 0;
      mn = mbit ? fminf(mn, e) : mn;
      mx = mbit ? fmaxf(mx, e) : mx;
    }
    float pooled = andv ? mn : mx;
    if (mb8 == 0ull) pooled = 0.f;
    if (nodeOk) xbf[(long)node * DIN + 32 + condOfs + t * 16 + r15] = (__bf16)pooled;
  }
}

// ---------------------------------------------------------------------------
// Kernel 3: x[:, 160:288] = bitmap @ W_bm.T + b_bm   (K=1000, padded tail)
// ---------------------------------------------------------------------------
__global__ void bitmap_k(const float* __restrict__ bitmap, const __bf16* __restrict__ Wbm,
                         const float* __restrict__ bbm, __bf16* __restrict__ xbf) {
  const int wave = threadIdx.x >> 5;
  const int lane = threadIdx.x & 31;
  const int half = lane >> 4, r15 = lane & 15;
  const long tile   = (long)blockIdx.x * 8 + wave;  // 4096 rowTiles x 8 colTiles
  const int rowTile = (int)(tile >> 3);
  const int colTile = (int)(tile & 7);
  f32x8 acc = {0.f, 0.f, 0.f, 0.f, 0.f, 0.f, 0.f, 0.f};
  acc = gemm_tile_f32A(bitmap, 1000, (long)rowTile * 16, N_NODES, Wbm, 1000, colTile * 16, 1000, acc);
  const float bias = bbm[colTile * 16 + r15];
#pragma unroll
  for (int v = 0; v < 8; ++v) {
    const int row = rowTile * 16 + v + 8 * half;
    if (row < N_NODES)
      xbf[(long)row * DIN + 160 + colTile * 16 + r15] = (__bf16)(acc[v] + bias);
  }
}

// ---------------------------------------------------------------------------
// Kernel 4: xg = x @ W_ih.T + (b_ih + b_hh)   (N x 1024, K=288 -> 9 WMMAs/tile)
// ---------------------------------------------------------------------------
__global__ void xg_gemm_k(const __bf16* __restrict__ xbf, const __bf16* __restrict__ Wih,
                          const float* __restrict__ bih, const float* __restrict__ bhh,
                          float* __restrict__ xg) {
  const int wave = threadIdx.x >> 5;
  const int lane = threadIdx.x & 31;
  const int half = lane >> 4, r15 = lane & 15;
  const long tile   = (long)blockIdx.x * 8 + wave;  // 4096 rowTiles x 64 colTiles
  const int rowTile = (int)(tile >> 6);
  const int colTile = (int)(tile & 63);
  f32x8 acc = {0.f, 0.f, 0.f, 0.f, 0.f, 0.f, 0.f, 0.f};
  acc = gemm_tile_bf16A(xbf, DIN, (long)rowTile * 16, N_NODES, Wih, DIN, colTile * 16, DIN, acc);
  const int col = colTile * 16 + r15;
  const float bias = bih[col] + bhh[col];
#pragma unroll
  for (int v = 0; v < 8; ++v) {
    const int row = rowTile * 16 + v + 8 * half;
    if (row < N_NODES)
      xg[(long)row * G4 + col] = acc[v] + bias;
  }
}

// ---------------------------------------------------------------------------
// Kernel 5: leaf level (c0 = h0 = 0): pure gate nonlinearity from xg
// ---------------------------------------------------------------------------
__global__ void leaf_k(const float* __restrict__ xg, float* __restrict__ H, float* __restrict__ C) {
  const long i = (long)blockIdx.x * blockDim.x + threadIdx.x;
  if (i >= 32768L * 256) return;
  const long node = i >> 8;
  const int  j    = (int)(i & 255);
  const float gi = xg[node * G4 + j];
  const float gg = xg[node * G4 + 512 + j];
  const float go = xg[node * G4 + 768 + j];
  const float c  = sigmoidf_(gi) * tanhf(gg);
  const float h  = sigmoidf_(go) * tanhf(c);
  C[node * HIDN + j] = c;
  H[node * HIDN + j] = h;
}

// ---------------------------------------------------------------------------
// Kernel 6: one internal tree level.  Block = 16 nodes.
// Phase 1: h0 = 0.5*(H[l]+H[r]) -> LDS bf16 (16B aligned for ds_load_b128);
//          c0 -> LDS f32.
// Phase 2: each wave owns j-tiles grp, grp+8; computes the four gate column
// tiles of  g = xg + h0 @ W_hh.T  via WMMA, then fuses the LSTM gates.
// ---------------------------------------------------------------------------
__global__ void level_k(const float* __restrict__ xg, float* __restrict__ H, float* __restrict__ C,
                        const __bf16* __restrict__ Whh, int start, int count) {
  __shared__ __align__(16) __bf16 h0s[16 * 256];
  __shared__ __align__(16) float  c0s[16 * 256];
  const int tileBase = blockIdx.x * 16;

  for (int i = threadIdx.x; i < 16 * 256; i += 256) {
    const int ln = i >> 8;
    const int j  = i & 255;
    const int node = start + tileBase + ln;
    float h0 = 0.f, c0 = 0.f;
    if (node < start + count) {
      const long idx = 2L * node - N_NODES;   // children: idx-1, idx
      h0 = 0.5f * (H[idx * HIDN + j] + H[(idx - 1) * HIDN + j]);
      c0 = 0.5f * (C[idx * HIDN + j] + C[(idx - 1) * HIDN + j]);
    }
    h0s[ln * 256 + j] = (__bf16)h0;
    c0s[ln * 256 + j] = c0;
  }
  __syncthreads();

  const int wave = threadIdx.x >> 5;
  const int lane = threadIdx.x & 31;
  const int half = lane >> 4, r15 = lane & 15;

  for (int grp = wave; grp < 16; grp += 8) {
    const int jcol = grp * 16 + r15;
    f32x8 acc[4];
#pragma unroll
    for (int gate = 0; gate < 4; ++gate) {
#pragma unroll
      for (int v = 0; v < 8; ++v) {
        const int ln = v + 8 * half;
        int node = start + tileBase + ln;
        if (node >= start + count) node = start + count - 1;
        acc[gate][v] = xg[(long)node * G4 + gate * 256 + jcol];
      }
      acc[gate] = gemm_tile_bf16A(h0s, 256, 0, 16, Whh, 256, gate * 256 + grp * 16, 256, acc[gate]);
    }
#pragma unroll
    for (int v = 0; v < 8; ++v) {
      const int ln = v + 8 * half;
      const int node = start + tileBase + ln;
      if (node < start + count) {
        const float gi = acc[0][v], gf = acc[1][v], gg = acc[2][v], go = acc[3][v];
        const float c0 = c0s[ln * 256 + jcol];
        const float c  = sigmoidf_(gf) * c0 + sigmoidf_(gi) * tanhf(gg);
        const float h  = sigmoidf_(go) * tanhf(c);
        C[(long)node * HIDN + jcol] = c;
        H[(long)node * HIDN + jcol] = h;
      }
    }
  }
}

// ---------------------------------------------------------------------------
// Kernel 7: the two tiny MLP heads on the root hidden state (1 block, VALU)
// ---------------------------------------------------------------------------
__global__ void head_k(const float* __restrict__ H,
                       const float* W2a, const float* b2a, const float* W3a, const float* b3a,
                       const float* Woa, const float* boa,
                       const float* W2b, const float* b2b, const float* W3b, const float* b3b,
                       const float* Wob, const float* bob,
                       float* out) {
  __shared__ float hs[256];
  __shared__ float t1[128];
  __shared__ float t2[128];
  __shared__ float red[128];
  const int tid = threadIdx.x;  // 128 threads
  for (int i = tid; i < 256; i += 128) hs[i] = H[(long)(N_NODES - 1) * HIDN + i];
  __syncthreads();
  for (int head = 0; head < 2; ++head) {
    const float* W2 = head ? W2b : W2a; const float* b2 = head ? b2b : b2a;
    const float* W3 = head ? W3b : W3a; const float* b3 = head ? b3b : b3a;
    const float* Wo = head ? Wob : Woa; const float* bo = head ? bob : boa;
    float a = b2[tid];
    for (int k = 0; k < 256; ++k) a += W2[tid * 256 + k] * hs[k];
    t1[tid] = fmaxf(a, 0.f);
    __syncthreads();
    float b = b3[tid];
    for (int k = 0; k < 128; ++k) b += W3[tid * 128 + k] * t1[k];
    t2[tid] = fmaxf(b, 0.f);
    __syncthreads();
    red[tid] = Wo[tid] * t2[tid];
    __syncthreads();
    for (int s = 64; s > 0; s >>= 1) {
      if (tid < s) red[tid] += red[tid + s];
      __syncthreads();
    }
    if (tid == 0) out[head] = sigmoidf_(red[0] + bo[0]);
    __syncthreads();
  }
}

// ---------------------------------------------------------------------------
extern "C" void kernel_launch(void* const* d_in, const int* in_sizes, int n_in,
                              void* d_out, int out_size, void* d_ws, size_t ws_size,
                              hipStream_t stream) {
  (void)in_sizes; (void)n_in; (void)out_size; (void)ws_size;

  const float* op_vec = (const float*)d_in[0];
  const float* c1p    = (const float*)d_in[1];
  const unsigned char* c1m = (const unsigned char*)d_in[2];
  const unsigned char* c1a = (const unsigned char*)d_in[3];
  const float* c2p    = (const float*)d_in[4];
  const unsigned char* c2m = (const unsigned char*)d_in[5];
  const unsigned char* c2a = (const unsigned char*)d_in[6];
  const float* bitmap = (const float*)d_in[7];
  const float* W_op   = (const float*)d_in[8];
  const float* b_op   = (const float*)d_in[9];
  const float* W_pred = (const float*)d_in[10];
  const float* b_pred = (const float*)d_in[11];
  const float* W_bm   = (const float*)d_in[12];
  const float* b_bm   = (const float*)d_in[13];
  const float* W_ih   = (const float*)d_in[14];
  const float* b_ih   = (const float*)d_in[15];
  const float* W_hh   = (const float*)d_in[16];
  const float* b_hh   = (const float*)d_in[17];

  // workspace carve-out (256B-aligned segments)
  char* p = (char*)d_ws;
  auto carve = [&](size_t bytes) -> void* {
    void* r = (void*)p;
    p += (bytes + 255) & ~(size_t)255;
    return r;
  };
  __bf16* xbf = (__bf16*)carve((size_t)N_NODES * DIN * sizeof(__bf16));
  float*  xg  = (float*) carve((size_t)N_NODES * G4 * sizeof(float));
  float*  H   = (float*) carve((size_t)N_NODES * HIDN * sizeof(float));
  float*  C   = (float*) carve((size_t)N_NODES * HIDN * sizeof(float));
  __bf16* Wpb = (__bf16*)carve((size_t)64 * 64 * sizeof(__bf16));
  __bf16* Wbb = (__bf16*)carve((size_t)128 * 1000 * sizeof(__bf16));
  __bf16* Wib = (__bf16*)carve((size_t)1024 * 288 * sizeof(__bf16));
  __bf16* Whb = (__bf16*)carve((size_t)1024 * 256 * sizeof(__bf16));

  // 0) weight conversion (largest segment 1024*288 = 294912 elements)
  cvt_weights_k<<<(294912 + 255) / 256, 256, 0, stream>>>(W_pred, W_bm, W_ih, W_hh,
                                                          Wpb, Wbb, Wib, Whb);
  // 1) featurizer
  op_feat_k<<<((long)N_NODES * 32 + 255) / 256, 256, 0, stream>>>(op_vec, W_op, b_op, xbf);
  pred_pool_k<<<4096, 256, 0, stream>>>(c1p, c1m, c1a, Wpb, b_pred, xbf, 0);
  pred_pool_k<<<4096, 256, 0, stream>>>(c2p, c2m, c2a, Wpb, b_pred, xbf, 64);
  bitmap_k<<<4096, 256, 0, stream>>>(bitmap, Wbb, b_bm, xbf);
  // 2) input-gate pre-activations for all nodes
  xg_gemm_k<<<32768, 256, 0, stream>>>(xbf, Wib, b_ih, b_hh, xg);
  // 3) tree LSTM, bottom-up
  leaf_k<<<32768, 256, 0, stream>>>(xg, H, C);
  for (int d = 14; d >= 0; --d) {
    const int count = 1 << d;
    const int start = N_NODES - (2 << d) + 1;
    const int blocks = (count + 15) / 16;
    level_k<<<blocks, 256, 0, stream>>>(xg, H, C, Whb, start, count);
  }
  // 4) heads on root hidden state
  head_k<<<1, 128, 0, stream>>>(H,
                                (const float*)d_in[18], (const float*)d_in[19],
                                (const float*)d_in[20], (const float*)d_in[21],
                                (const float*)d_in[22], (const float*)d_in[23],
                                (const float*)d_in[24], (const float*)d_in[25],
                                (const float*)d_in[26], (const float*)d_in[27],
                                (const float*)d_in[28], (const float*)d_in[29],
                                (float*)d_out);
}
